// BAZ_Network_16054587752902
// MI455X (gfx1250) — matrix operand order repaired and checked by reading, not compiled, not run on hardware
//
#include <hip/hip_runtime.h>
#include <hip/hip_bf16.h>
#include <math.h>

// ---------------------------------------------------------------------------
// Types for CDNA5 WMMA (wave32): v_wmma_f32_16x16x32_bf16
// ---------------------------------------------------------------------------
typedef __attribute__((ext_vector_type(16))) __bf16 bf16x16;
typedef __attribute__((ext_vector_type(8)))  __bf16 bf16x8;
typedef __attribute__((ext_vector_type(8)))  float  f32x8;

#define NBATCH   1024
#define CIN      3
#define LIN      6000
#define FLATDIM  7640              /* 20*375 + 20*7 */
#define FLATPAD  7648              /* padded to 32-multiple for WMMA K */
#define HID      100
#define HIDPAD   112               /* padded to 16-multiple */

// ---------------------------------------------------------------------------
// Kernel 0: init global maxima (ws is poisoned; must re-init every call)
// ---------------------------------------------------------------------------
__global__ void init_gmax_kernel(float* gmax) {
    if (threadIdx.x < 2) gmax[threadIdx.x] = 0.0f;
}

__device__ __forceinline__ void atomic_max_pos_f32(float* addr, float v) {
    // valid for non-negative floats: IEEE ordering == unsigned bit ordering
    atomicMax(reinterpret_cast<unsigned int*>(addr), __float_as_uint(fmaxf(v, 0.0f)));
}

// ---------------------------------------------------------------------------
// Prep kernels: f32 -> bf16 conversions / zero-padded weight repack
// ---------------------------------------------------------------------------
__global__ __launch_bounds__(256) void f32_to_bf16_kernel(const float* __restrict__ src,
                                                          __bf16* __restrict__ dst,
                                                          int count) {
    int i = blockIdx.x * 256 + threadIdx.x;
    if (i < count) dst[i] = (__bf16)src[i];
}

// wl0 (100 x 7640 f32) -> (112 x 7648 bf16), zero padded
__global__ __launch_bounds__(256) void pad_wl0_kernel(const float* __restrict__ wl0,
                                                      __bf16* __restrict__ wl0b) {
    int i = blockIdx.x * 256 + threadIdx.x;
    if (i >= HIDPAD * FLATPAD) return;
    int hu = i / FLATPAD, k = i % FLATPAD;
    float v = (hu < HID && k < FLATDIM) ? wl0[hu * FLATDIM + k] : 0.0f;
    wl0b[i] = (__bf16)v;
}

// ---------------------------------------------------------------------------
// Kernel 1: per-sample covariance + closed-form 3x3 symmetric eig
//   stats per sample (21 f32): cov[9], vals[3] ascending, vecs[9] row-major
//   gmax[0] = max|cov| over batch, gmax[1] = max eigenvalue over batch
// ---------------------------------------------------------------------------
__device__ __forceinline__ void cross3(const float a[3], const float b[3], float o[3]) {
    o[0] = a[1]*b[2] - a[2]*b[1];
    o[1] = a[2]*b[0] - a[0]*b[2];
    o[2] = a[0]*b[1] - a[1]*b[0];
}

__global__ __launch_bounds__(256) void cov_eig_kernel(const float* __restrict__ x,
                                                      float* __restrict__ stats,
                                                      float* __restrict__ gmax) {
    const int n   = blockIdx.x;
    const int tid = threadIdx.x;
    const float* xb = x + (long)n * CIN * LIN;

    float s[3]  = {0.f, 0.f, 0.f};
    float pr[6] = {0.f, 0.f, 0.f, 0.f, 0.f, 0.f};  // 00,01,02,11,12,22
    for (int l = tid; l < LIN; l += 256) {
        float a = xb[l], b = xb[LIN + l], c = xb[2 * LIN + l];
        s[0] += a; s[1] += b; s[2] += c;
        pr[0] += a * a; pr[1] += a * b; pr[2] += a * c;
        pr[3] += b * b; pr[4] += b * c; pr[5] += c * c;
    }

    __shared__ float red[9][256];
    red[0][tid] = s[0]; red[1][tid] = s[1]; red[2][tid] = s[2];
    #pragma unroll
    for (int j = 0; j < 6; ++j) red[3 + j][tid] = pr[j];
    __syncthreads();
    for (int st = 128; st > 0; st >>= 1) {
        if (tid < st)
            #pragma unroll
            for (int j = 0; j < 9; ++j) red[j][tid] += red[j][tid + st];
        __syncthreads();
    }

    if (tid == 0) {
        const float L = (float)LIN;
        float m0 = red[0][0] / L, m1 = red[1][0] / L, m2 = red[2][0] / L;
        float inv = 1.0f / (L - 1.0f);
        float c00 = (red[3][0] - L * m0 * m0) * inv;
        float c01 = (red[4][0] - L * m0 * m1) * inv;
        float c02 = (red[5][0] - L * m0 * m2) * inv;
        float c11 = (red[6][0] - L * m1 * m1) * inv;
        float c12 = (red[7][0] - L * m1 * m2) * inv;
        float c22 = (red[8][0] - L * m2 * m2) * inv;

        float* st9 = stats + (long)n * 21;
        st9[0] = c00; st9[1] = c01; st9[2] = c02;
        st9[3] = c01; st9[4] = c11; st9[5] = c12;
        st9[6] = c02; st9[7] = c12; st9[8] = c22;

        float p1 = c01 * c01 + c02 * c02 + c12 * c12;
        float q  = (c00 + c11 + c22) * (1.0f / 3.0f);
        float d0 = c00 - q, d1 = c11 - q, d2 = c22 - q;
        float p2 = d0 * d0 + d1 * d1 + d2 * d2 + 2.0f * p1;
        float p  = sqrtf(p2 * (1.0f / 6.0f));
        float ev0, ev1, ev2;
        if (p > 1e-30f) {
            float ip = 1.0f / p;
            float b00 = d0 * ip, b11 = d1 * ip, b22 = d2 * ip;
            float b01 = c01 * ip, b02 = c02 * ip, b12 = c12 * ip;
            float detB = b00 * (b11 * b22 - b12 * b12)
                       - b01 * (b01 * b22 - b12 * b02)
                       + b02 * (b01 * b12 - b11 * b02);
            float r = fminf(1.0f, fmaxf(-1.0f, 0.5f * detB));
            float phi = acosf(r) * (1.0f / 3.0f);
            float eA = q + 2.0f * p * cosf(phi);                        // largest
            float eC = q + 2.0f * p * cosf(phi + 2.0943951023931953f);  // smallest
            float eB = 3.0f * q - eA - eC;
            ev0 = eC; ev1 = eB; ev2 = eA;
        } else {
            ev0 = ev1 = ev2 = q;
        }
        st9[9] = ev0; st9[10] = ev1; st9[11] = ev2;

        float evs[3] = {ev0, ev1, ev2};
        float vecs[3][3];
        #pragma unroll
        for (int i = 0; i < 3; ++i) {
            float lam = evs[i];
            float r0[3] = {c00 - lam, c01, c02};
            float r1[3] = {c01, c11 - lam, c12};
            float r2[3] = {c02, c12, c22 - lam};
            float cA[3], cB[3], cC[3];
            cross3(r0, r1, cA); cross3(r0, r2, cB); cross3(r1, r2, cC);
            float nA = cA[0]*cA[0] + cA[1]*cA[1] + cA[2]*cA[2];
            float nB = cB[0]*cB[0] + cB[1]*cB[1] + cB[2]*cB[2];
            float nC = cC[0]*cC[0] + cC[1]*cC[1] + cC[2]*cC[2];
            float* best = cA; float nb = nA;
            if (nB > nb) { best = cB; nb = nB; }
            if (nC > nb) { best = cC; nb = nC; }
            if (nb > 1e-30f) {
                float inorm = rsqrtf(nb);
                vecs[0][i] = best[0] * inorm;
                vecs[1][i] = best[1] * inorm;
                vecs[2][i] = best[2] * inorm;
            } else {
                vecs[0][i] = (i == 0); vecs[1][i] = (i == 1); vecs[2][i] = (i == 2);
            }
        }
        #pragma unroll
        for (int r = 0; r < 3; ++r)
            #pragma unroll
            for (int i = 0; i < 3; ++i) st9[12 + r * 3 + i] = vecs[r][i];

        float covmax = 0.f;
        covmax = fmaxf(covmax, fabsf(c00)); covmax = fmaxf(covmax, fabsf(c01));
        covmax = fmaxf(covmax, fabsf(c02)); covmax = fmaxf(covmax, fabsf(c11));
        covmax = fmaxf(covmax, fabsf(c12)); covmax = fmaxf(covmax, fabsf(c22));
        atomic_max_pos_f32(&gmax[0], covmax);
        atomic_max_pos_f32(&gmax[1], ev2);
    }
}

// ---------------------------------------------------------------------------
// Kernel 2: normalized feats (N,3,7) -> 1x1 conv (3->20) + ReLU ->
//   bf16 into concatb[n*7648 + 7500 .. 7639]; also zero the pad [7640..7647]
// ---------------------------------------------------------------------------
__global__ __launch_bounds__(64) void feats_kernel(const float* __restrict__ stats,
                                                   const float* __restrict__ gmax,
                                                   const float* __restrict__ wc,
                                                   const float* __restrict__ bc,
                                                   __bf16* __restrict__ concatb) {
    const int n   = blockIdx.x;
    const int tid = threadIdx.x;
    __shared__ float f[3][7];
    const float* st = stats + (long)n * 21;
    if (tid < 21) {
        int c = tid / 7, j = tid % 7;
        float v;
        if (j < 3)       v = st[c * 3 + j] / gmax[0];
        else if (j == 3) v = st[9 + c]     / gmax[1];
        else             v = st[12 + c * 3 + (j - 4)];
        f[c][j] = v;
    }
    __syncthreads();
    __bf16* row = concatb + (long)n * FLATPAD;
    for (int i = tid; i < 140; i += 64) {
        int oc = i / 7, j = i % 7;
        float acc = bc[oc];
        #pragma unroll
        for (int c = 0; c < 3; ++c) acc += f[c][j] * wc[oc * 3 + c];
        row[7500 + oc * 7 + j] = (__bf16)fmaxf(acc, 0.0f);
    }
    if (tid < FLATPAD - FLATDIM) row[FLATDIM + tid] = (__bf16)0.0f;
}

// ---------------------------------------------------------------------------
// Kernel 3 (templated): conv1d(K=3,SAME) + bias + ReLU + maxpool2 as implicit
// GEMM on v_wmma_f32_16x16x32_bf16, all-bf16 data path.
//   Phase 1: stage input tile (halo, zero-padded channels up to ICpad) + weights
//   Phase 2: build K-major im2col tile sColT[j][k] in LDS (zero-padded to Kpad)
//   Phase 3: wave tiles -> A/B fragments are contiguous 16B LDS vector loads
//            (A and B layout: K runs {8g..8g+7} and {16+8g..23+8g})
//   Epilogue: bias+ReLU, pool2 via shfl_xor lane pairing, bf16 store.
// ---------------------------------------------------------------------------
template <int IC, int OC, int Lin, int TILE>
__global__ __launch_bounds__(256) void conv_relu_pool_wmma(const __bf16* __restrict__ in,
                                                           const float* __restrict__ w,
                                                           const float* __restrict__ bias,
                                                           __bf16* __restrict__ out,
                                                           long sampleStride) {
    constexpr int KDIM  = IC * 3;
    constexpr int KC    = (KDIM + 31) / 32;
    constexpr int Kpad  = KC * 32;
    constexpr int ICpad = (Kpad + 2) / 3;          // channels covering Kpad im2col rows
    constexpr int MT    = (OC + 15) / 16;
    constexpr int Mpad  = MT * 16;
    constexpr int HALO  = TILE + 2;

    const int n    = blockIdx.y;
    const int p0   = blockIdx.x * TILE;
    const int tid  = threadIdx.x;
    const int lane = tid & 31;
    const int wave = tid >> 5;
    const int g    = lane >> 4;
    const int lm   = lane & 15;

    __shared__ __align__(16) __bf16 sIn[ICpad * HALO];
    __shared__ __align__(16) __bf16 sColT[TILE * Kpad];   // [j][k]
    __shared__ __align__(16) __bf16 sW[Mpad * Kpad];      // [m][k]
    __shared__ float sBias[Mpad];

    // Phase 1: weights (bf16, zero-padded) + input tile with halo
    for (int i = tid; i < Mpad * Kpad; i += 256) {
        int m = i / Kpad, k = i % Kpad;
        float v = (m < OC && k < KDIM) ? w[m * KDIM + k] : 0.0f;
        sW[i] = (__bf16)v;
    }
    for (int i = tid; i < Mpad; i += 256) sBias[i] = (i < OC) ? bias[i] : 0.0f;
    const __bf16* inN = in + (long)n * IC * Lin;
    for (int i = tid; i < ICpad * HALO; i += 256) {
        int c = i / HALO, j = i % HALO;
        int p = p0 - 1 + j;
        __bf16 v = (__bf16)0.0f;
        if (c < IC && p >= 0 && p < Lin) v = inN[c * Lin + p];
        sIn[i] = v;
    }
    __syncthreads();

    // Phase 2: K-major im2col tile
    for (int i = tid; i < TILE * Kpad; i += 256) {
        int j = i / Kpad, k = i % Kpad;
        int c = k / 3, r = k - 3 * c;
        sColT[i] = sIn[c * HALO + j + r];   // srcpos = p0 + j + r - 1
    }
    __syncthreads();

    // Phase 3: WMMA tiles
    for (int wi = wave; wi < MT * (TILE / 16); wi += 8) {
        const int mt = wi / (TILE / 16);
        const int pt = wi % (TILE / 16);
        const __bf16* rowA = &sW[(mt * 16 + lm) * Kpad + 8 * g];
        const __bf16* colB = &sColT[(pt * 16 + lm) * Kpad + 8 * g];
        f32x8 acc = {};
        #pragma unroll
        for (int kc = 0; kc < KC; ++kc) {
            bf16x8 alo = *(const bf16x8*)(rowA + kc * 32);
            bf16x8 ahi = *(const bf16x8*)(rowA + kc * 32 + 16);
            bf16x8 blo = *(const bf16x8*)(colB + kc * 32);
            bf16x8 bhi = *(const bf16x8*)(colB + kc * 32 + 16);
            bf16x16 a = __builtin_shufflevector(alo, ahi, 0,1,2,3,4,5,6,7,8,9,10,11,12,13,14,15);
            bf16x16 b = __builtin_shufflevector(blo, bhi, 0,1,2,3,4,5,6,7,8,9,10,11,12,13,14,15);
            acc = __builtin_amdgcn_wmma_f32_16x16x32_bf16(false, a, false, b,
                                                          (short)0, acc, false, false);
        }
        // Epilogue: bias + ReLU + pool2 (lane pairs share oc; positions p, p+1)
        const int pos = p0 + pt * 16 + lm;
        #pragma unroll
        for (int r = 0; r < 8; ++r) {
            int oc = mt * 16 + r + 8 * g;             // D layout: M = r + 8*g, N = lm
            float v = fmaxf(acc[r] + sBias[oc], 0.0f);
            float pooled = fmaxf(v, __shfl_xor(v, 1, 32));
            if (((lm & 1) == 0) && oc < OC && pos < Lin) {
                out[(long)n * sampleStride + (long)oc * (Lin / 2) + (pos >> 1)] = (__bf16)pooled;
            }
        }
    }
}

// ---------------------------------------------------------------------------
// Kernel 4: FC1  hid = relu(concatb(1024x7648 bf16) @ wl0b^T(7648x112 bf16) + bl0)
//   Fully padded operands -> zero branches: 4 aligned global vector loads + 1
//   wmma per 32-K chunk.
// ---------------------------------------------------------------------------
__global__ __launch_bounds__(256) void fc1_wmma(const __bf16* __restrict__ act,
                                                const __bf16* __restrict__ wl0b,
                                                const float* __restrict__ bl0,
                                                float* __restrict__ hid) {
    const int tid  = threadIdx.x;
    const int lane = tid & 31;
    const int wave = tid >> 5;
    const int g    = lane >> 4;
    const int lm   = lane & 15;
    const int s0   = blockIdx.x * 16;
    constexpr int KC = FLATPAD / 32;   // 239

    for (int nt = wave; nt < HIDPAD / 16; nt += 8) {   // 7 tiles, 8 waves
        const int hu = nt * 16 + lm;
        const __bf16* rowA = act + (long)(s0 + lm) * FLATPAD + 8 * g;
        const __bf16* rowB = wl0b + (long)hu * FLATPAD + 8 * g;
        f32x8 acc = {};
        for (int kc = 0; kc < KC; ++kc) {
            bf16x8 alo = *(const bf16x8*)(rowA + kc * 32);
            bf16x8 ahi = *(const bf16x8*)(rowA + kc * 32 + 16);
            bf16x8 blo = *(const bf16x8*)(rowB + kc * 32);
            bf16x8 bhi = *(const bf16x8*)(rowB + kc * 32 + 16);
            bf16x16 a = __builtin_shufflevector(alo, ahi, 0,1,2,3,4,5,6,7,8,9,10,11,12,13,14,15);
            bf16x16 b = __builtin_shufflevector(blo, bhi, 0,1,2,3,4,5,6,7,8,9,10,11,12,13,14,15);
            acc = __builtin_amdgcn_wmma_f32_16x16x32_bf16(false, a, false, b,
                                                          (short)0, acc, false, false);
        }
        if (hu < HID) {
            float bv = bl0[hu];
            #pragma unroll
            for (int r = 0; r < 8; ++r) {
                int s = s0 + r + 8 * g;
                hid[(long)s * HID + hu] = fmaxf(acc[r] + bv, 0.0f);
            }
        }
    }
}

// ---------------------------------------------------------------------------
// Kernel 5: FC2 (100 -> 2), outputs split as [out[:,0] ; out[:,1]]
// ---------------------------------------------------------------------------
__global__ __launch_bounds__(256) void fc2_kernel(const float* __restrict__ hid,
                                                  const float* __restrict__ wl1,
                                                  const float* __restrict__ bl1,
                                                  float* __restrict__ out) {
    int idx = blockIdx.x * blockDim.x + threadIdx.x;
    if (idx >= 2 * NBATCH) return;
    int n = idx >> 1, c = idx & 1;
    const float* h = hid + (long)n * HID;
    const float* w = wl1 + c * HID;
    float acc = bl1[c];
    #pragma unroll 4
    for (int k = 0; k < HID; ++k) acc += h[k] * w[k];
    out[c * NBATCH + n] = acc;
}

// ---------------------------------------------------------------------------
// Launch
// ---------------------------------------------------------------------------
extern "C" void kernel_launch(void* const* d_in, const int* in_sizes, int n_in,
                              void* d_out, int out_size, void* d_ws, size_t ws_size,
                              hipStream_t stream) {
    const float* x   = (const float*)d_in[0];
    const float* w0  = (const float*)d_in[1];
    const float* b0  = (const float*)d_in[2];
    const float* w1  = (const float*)d_in[3];
    const float* b1  = (const float*)d_in[4];
    const float* w2  = (const float*)d_in[5];
    const float* b2  = (const float*)d_in[6];
    const float* w3  = (const float*)d_in[7];
    const float* b3  = (const float*)d_in[8];
    const float* wc  = (const float*)d_in[9];
    const float* bc  = (const float*)d_in[10];
    const float* wl0 = (const float*)d_in[11];
    const float* bl0 = (const float*)d_in[12];
    const float* wl1 = (const float*)d_in[13];
    const float* bl1 = (const float*)d_in[14];
    float* out = (float*)d_out;

    char* ws = (char*)d_ws;
    size_t off = 0;
    auto walloc = [&](size_t bytes) -> void* {
        void* p = ws + off;
        off = (off + bytes + 255) & ~(size_t)255;
        return p;
    };
    float*  stats   = (float*)walloc((size_t)NBATCH * 21 * sizeof(float));
    float*  gmax    = (float*)walloc(2 * sizeof(float));
    float*  hid     = (float*)walloc((size_t)NBATCH * HID * sizeof(float));
    __bf16* concatb = (__bf16*)walloc((size_t)NBATCH * FLATPAD * sizeof(__bf16));
    __bf16* xb      = (__bf16*)walloc((size_t)NBATCH * CIN * LIN * sizeof(__bf16));
    __bf16* wl0b    = (__bf16*)walloc((size_t)HIDPAD * FLATPAD * sizeof(__bf16));
    __bf16* bufA    = (__bf16*)walloc((size_t)NBATCH * 20 * 3000 * sizeof(__bf16)); // conv0/conv2 out
    __bf16* bufB    = (__bf16*)walloc((size_t)NBATCH * 32 * 1500 * sizeof(__bf16)); // conv1 out

    // prep
    init_gmax_kernel<<<1, 32, 0, stream>>>(gmax);
    {
        int cnt = NBATCH * CIN * LIN;
        f32_to_bf16_kernel<<<(cnt + 255) / 256, 256, 0, stream>>>(x, xb, cnt);
    }
    pad_wl0_kernel<<<(HIDPAD * FLATPAD + 255) / 256, 256, 0, stream>>>(wl0, wl0b);

    // covariance features branch
    cov_eig_kernel<<<NBATCH, 256, 0, stream>>>(x, stats, gmax);
    feats_kernel<<<NBATCH, 64, 0, stream>>>(stats, gmax, wc, bc, concatb);

    // conv chain (bf16 everywhere), conv3 writes straight into concat layout
    conv_relu_pool_wmma<3, 20, 6000, 256><<<dim3(24, NBATCH), 256, 0, stream>>>(xb,   w0, b0, bufA,    (long)20 * 3000);
    conv_relu_pool_wmma<20, 32, 3000, 128><<<dim3(24, NBATCH), 256, 0, stream>>>(bufA, w1, b1, bufB,   (long)32 * 1500);
    conv_relu_pool_wmma<32, 64, 1500, 128><<<dim3(12, NBATCH), 256, 0, stream>>>(bufB, w2, b2, bufA,   (long)64 * 750);
    conv_relu_pool_wmma<64, 20, 750,  64><<<dim3(12, NBATCH), 256, 0, stream>>>(bufA, w3, b3, concatb, (long)FLATPAD);

    // head
    fc1_wmma<<<NBATCH / 16, 256, 0, stream>>>(concatb, wl0b, bl0, hid);
    fc2_kernel<<<(2 * NBATCH + 255) / 256, 256, 0, stream>>>(hid, wl1, bl1, out);
}
